// GNNEncoder4_58016418234919
// MI455X (gfx1250) — compile-verified, matrix operand order checked by reading, not run on hardware
//
#include <hip/hip_runtime.h>
#include <hip/hip_bf16.h>

typedef _Float16 half_t;
typedef __attribute__((ext_vector_type(16))) _Float16 v16h;
typedef __attribute__((ext_vector_type(8)))  _Float16 v8h;
typedef __attribute__((ext_vector_type(4)))  _Float16 v4h;
typedef __attribute__((ext_vector_type(8)))  float    v8f;

#define BM    128   // rows per block
#define NOUT  128   // fixed output width (H*C)
#define TK    32    // K step per WMMA
#define ASTR  40    // LDS row stride in halves (80B: 16B-aligned, conflict-skewed)
#define BUFH  (2 * BM * ASTR)   // halves per double-buffer slot (A tile + B tile)

// ---------------------------------------------------------------------------
// CDNA5 async global->LDS copy (ASYNCcnt) via inline asm
// ---------------------------------------------------------------------------
__device__ __forceinline__ void async_copy16(unsigned lds_byte, const half_t* gsrc) {
    asm volatile("global_load_async_to_lds_b128 %0, %1, off"
                 :: "v"(lds_byte), "v"(gsrc)
                 : "memory");
}
__device__ __forceinline__ void wait_async0() {
    asm volatile("s_wait_asynccnt 0x0" ::: "memory");
}

// ---------------------------------------------------------------------------
// fp32 -> f16 bulk convert (4 elems / thread), n % 4 == 0 for all our buffers
// ---------------------------------------------------------------------------
__global__ void f32_to_f16(const float* __restrict__ src, half_t* __restrict__ dst, long n) {
    long base = ((long)blockIdx.x * blockDim.x + threadIdx.x) * 4;
    if (base >= n) return;
    float4 f = *(const float4*)(src + base);
    v4h h = { (half_t)f.x, (half_t)f.y, (half_t)f.z, (half_t)f.w };
    *(v4h*)(dst + base) = h;
}

// fp32 relu -> f16
__global__ void relu_to_f16(const float* __restrict__ src, half_t* __restrict__ dst, long n) {
    long base = ((long)blockIdx.x * blockDim.x + threadIdx.x) * 4;
    if (base >= n) return;
    float4 f = *(const float4*)(src + base);
    v4h h = { (half_t)fmaxf(f.x, 0.f), (half_t)fmaxf(f.y, 0.f),
              (half_t)fmaxf(f.z, 0.f), (half_t)fmaxf(f.w, 0.f) };
    *(v4h*)(dst + base) = h;
}

// ---------------------------------------------------------------------------
// Weight pre-transpose + fp32 -> f16 conversion: W[K,128] -> WT[128,K] (f16)
// ---------------------------------------------------------------------------
__global__ void transpose_to_f16(const float* __restrict__ W, half_t* __restrict__ WT, int K) {
    int idx = blockIdx.x * blockDim.x + threadIdx.x;
    if (idx >= K * 128) return;
    int n = idx & 127;
    int k = idx >> 7;
    WT[n * K + k] = (half_t)W[k * 128 + n];
}

// ---------------------------------------------------------------------------
// Y[M,128] = A[M,K](f16) @ W[K,128] + bias   (f16 WMMA, f32 accumulate)
// A f16 row-major, WT f16 [128][K] (pre-transposed), K % 32 == 0
// Block: 256 threads (8 waves). Double-buffered LDS tiles filled with
// global_load_async_to_lds_b128; wave computes 16 rows x 128 cols (8 accs).
// ---------------------------------------------------------------------------
__global__ __launch_bounds__(256)
void gemm128_wmma_async(const half_t* __restrict__ A, const half_t* __restrict__ WT,
                        const float* __restrict__ bias, float* __restrict__ Y,
                        int M, int K)
{
    __shared__ alignas(16) half_t smem[2 * BUFH];   // [buf][A 128x40 | B 128x40]

    const int t    = threadIdx.x;
    const int lane = t & 31;
    const int wave = t >> 5;            // 0..7
    const int m0   = blockIdx.x * BM;
    const unsigned smemBase = (unsigned)(uintptr_t)&smem[0];

    v8f acc[8];
#pragma unroll
    for (int i = 0; i < 8; ++i) acc[i] = (v8f)0.f;

    // issue async copies for K-tile `it` into buffer `b`
    auto issue_tile = [&](int it, int b) {
        const int k0 = it * TK;
#pragma unroll
        for (int u = 0; u < 2; ++u) {
            int c   = t + 256 * u;      // chunk 0..511 (16B each)
            int row = c >> 2;           // 0..127
            int seg = (c & 3) * 8;      // half offset within 32-K row
            int ar  = m0 + row; if (ar > M - 1) ar = M - 1;   // clamp (masked at epilogue)
            // A tile
            async_copy16(smemBase + 2u * (unsigned)(b * BUFH + row * ASTR + seg),
                         A + (size_t)ar * K + k0 + seg);
            // B tile (WT rows are output columns)
            async_copy16(smemBase + 2u * (unsigned)(b * BUFH + BM * ASTR + row * ASTR + seg),
                         WT + (size_t)row * K + k0 + seg);
        }
    };

    const int nT = K / TK;
    issue_tile(0, 0);
    wait_async0();
    __syncthreads();

    const int am = lane & 15;           // M (or N) within 16x16 tile
    const int kh = lane >> 4;           // K-half selector (ISA 16-bit layout)

    for (int it = 0; it < nT; ++it) {
        const int b = it & 1;
        if (it + 1 < nT) issue_tile(it + 1, b ^ 1);   // prefetch next tile

        const half_t* Ab = smem + b * BUFH;
        const half_t* Bb = Ab + BM * ASTR;

        const half_t* arow = &Ab[(wave * 16 + am) * ASTR + 8 * kh];
        v8h alo = *(const v8h*)arow;            // K = 8*kh .. +7
        v8h ahi = *(const v8h*)(arow + 16);     // K = 16+8*kh .. +7
        v16h afrag;
#pragma unroll
        for (int i = 0; i < 8; ++i) { afrag[i] = alo[i]; afrag[i + 8] = ahi[i]; }

#pragma unroll
        for (int nt = 0; nt < 8; ++nt) {
            const half_t* brow = &Bb[(nt * 16 + am) * ASTR + 8 * kh];
            v8h blo = *(const v8h*)brow;
            v8h bhi = *(const v8h*)(brow + 16);
            v16h bfrag;
#pragma unroll
            for (int i = 0; i < 8; ++i) { bfrag[i] = blo[i]; bfrag[i + 8] = bhi[i]; }
            acc[nt] = __builtin_amdgcn_wmma_f32_16x16x32_f16(
                false, afrag, false, bfrag, (short)0, acc[nt], false, false);
        }

        wait_async0();      // next buffer ready; this wave's ds reads already done
        __syncthreads();    // all waves done with buffer b before it is rewritten
    }

    // ---- epilogue: C/D layout -> fp32 rows, add bias ----
    const int nlo = lane & 15;
    const int mhi = (lane >> 4) * 8;
#pragma unroll
    for (int nt = 0; nt < 8; ++nt) {
        int col = nt * 16 + nlo;
        float bv = bias ? bias[col] : 0.f;
#pragma unroll
        for (int r = 0; r < 8; ++r) {
            int row = m0 + wave * 16 + r + mhi;
            if (row < M) Y[(size_t)row * NOUT + col] = acc[nt][r] + bv;
        }
    }
}

// ---------------------------------------------------------------------------
// Segment-softmax attention over edges
// ---------------------------------------------------------------------------
__global__ void init_stats(unsigned* __restrict__ mkey, float* __restrict__ ssum, int n) {
    int i = blockIdx.x * blockDim.x + threadIdx.x;
    if (i < n) { mkey[i] = 0u; ssum[i] = 0.f; }
}

__device__ __forceinline__ unsigned f2key(float f) {
    unsigned u = __float_as_uint(f);
    return (u & 0x80000000u) ? ~u : (u | 0x80000000u);
}
__device__ __forceinline__ float key2f(unsigned key) {
    unsigned u = (key & 0x80000000u) ? (key & 0x7fffffffu) : ~key;
    return __uint_as_float(u);
}

__global__ void alpha_kernel(const long long* __restrict__ srcI, const long long* __restrict__ dstI,
                             const float* __restrict__ q, const float* __restrict__ k,
                             const float* __restrict__ eemb, float* __restrict__ alpha,
                             unsigned* __restrict__ mkey, int nE)
{
    int idx = blockIdx.x * blockDim.x + threadIdx.x;
    if (idx >= nE * 4) return;
    int e = idx >> 2, h = idx & 3;
    int s = (int)srcI[e], d = (int)dstI[e];
    const float4* qp = (const float4*)(q    + (size_t)d * 128 + h * 32);
    const float4* kp = (const float4*)(k    + (size_t)s * 128 + h * 32);
    const float4* ep = (const float4*)(eemb + (size_t)e * 128 + h * 32);
    float acc = 0.f;
#pragma unroll
    for (int i = 0; i < 8; ++i) {
        float4 qv = qp[i], kv = kp[i], ev = ep[i];
        acc += qv.x * (kv.x + ev.x) + qv.y * (kv.y + ev.y)
             + qv.z * (kv.z + ev.z) + qv.w * (kv.w + ev.w);
    }
    acc *= 0.17677669529663687f;   // 1/sqrt(C), C=32
    alpha[idx] = acc;
    atomicMax(mkey + d * 4 + h, f2key(acc));
}

__global__ void expsum_kernel(const long long* __restrict__ dstI, float* __restrict__ alpha,
                              const unsigned* __restrict__ mkey, float* __restrict__ ssum, int nE)
{
    int idx = blockIdx.x * blockDim.x + threadIdx.x;
    if (idx >= nE * 4) return;
    int e = idx >> 2, h = idx & 3;
    int d = (int)dstI[e];
    float a = __expf(alpha[idx] - key2f(mkey[d * 4 + h]));
    alpha[idx] = a;
    atomicAdd(ssum + d * 4 + h, a);
}

__global__ void aggregate_kernel(const long long* __restrict__ srcI, const long long* __restrict__ dstI,
                                 const float* __restrict__ v, const float* __restrict__ eemb,
                                 const float* __restrict__ alpha, const float* __restrict__ ssum,
                                 float* __restrict__ out, int nE)
{
    int idx = blockIdx.x * blockDim.x + threadIdx.x;
    if (idx >= nE * 4) return;
    int e = idx >> 2, h = idx & 3;
    int s = (int)srcI[e], d = (int)dstI[e];
    float coef = alpha[idx] / fmaxf(ssum[d * 4 + h], 1e-16f);
    const float4* vp = (const float4*)(v    + (size_t)s * 128 + h * 32);
    const float4* ep = (const float4*)(eemb + (size_t)e * 128 + h * 32);
    float* op = out + (size_t)d * 128 + h * 32;
#pragma unroll
    for (int i = 0; i < 8; ++i) {
        float4 vv = vp[i], ev = ep[i];
        atomicAdd(op + 4 * i + 0, coef * (vv.x + ev.x));
        atomicAdd(op + 4 * i + 1, coef * (vv.y + ev.y));
        atomicAdd(op + 4 * i + 2, coef * (vv.z + ev.z));
        atomicAdd(op + 4 * i + 3, coef * (vv.w + ev.w));
    }
}

// ---------------------------------------------------------------------------
// Host orchestration
// ---------------------------------------------------------------------------
extern "C" void kernel_launch(void* const* d_in, const int* in_sizes, int n_in,
                              void* d_out, int out_size, void* d_ws, size_t ws_size,
                              hipStream_t stream)
{
    const float*     x   = (const float*)d_in[0];
    const long long* ei  = (const long long*)d_in[1];   // int64 [2, E]
    const float*     ea  = (const float*)d_in[2];
    const float* wq1 = (const float*)d_in[3];  const float* bq1 = (const float*)d_in[4];
    const float* wk1 = (const float*)d_in[5];  const float* bk1 = (const float*)d_in[6];
    const float* wv1 = (const float*)d_in[7];  const float* bv1 = (const float*)d_in[8];
    const float* we1 = (const float*)d_in[9];
    const float* ws1 = (const float*)d_in[10]; const float* bs1 = (const float*)d_in[11];
    const float* wq2 = (const float*)d_in[12]; const float* bq2 = (const float*)d_in[13];
    const float* wk2 = (const float*)d_in[14]; const float* bk2 = (const float*)d_in[15];
    const float* wv2 = (const float*)d_in[16]; const float* bv2 = (const float*)d_in[17];
    const float* we2 = (const float*)d_in[18];
    const float* ws2 = (const float*)d_in[19]; const float* bs2 = (const float*)d_in[20];

    const int Nn = in_sizes[0] / 256;     // 100000 nodes
    const int E  = in_sizes[2] / 64;      // 1600000 edges
    const long long* srcI = ei;
    const long long* dstI = ei + E;

    // ---- workspace carve-up ----
    char* base = (char*)d_ws;
    size_t off = 0;
    auto alloc = [&](size_t bytes) -> char* {
        char* p = base + off;
        off = (off + bytes + 255) & ~(size_t)255;
        return p;
    };
    half_t* wq1T = (half_t*)alloc((size_t)128 * 256 * 2);
    half_t* wk1T = (half_t*)alloc((size_t)128 * 256 * 2);
    half_t* wv1T = (half_t*)alloc((size_t)128 * 256 * 2);
    half_t* ws1T = (half_t*)alloc((size_t)128 * 256 * 2);
    half_t* we1T = (half_t*)alloc((size_t)128 * 64 * 2);
    half_t* wq2T = (half_t*)alloc((size_t)128 * 128 * 2);
    half_t* wk2T = (half_t*)alloc((size_t)128 * 128 * 2);
    half_t* wv2T = (half_t*)alloc((size_t)128 * 128 * 2);
    half_t* ws2T = (half_t*)alloc((size_t)128 * 128 * 2);
    half_t* we2T = (half_t*)alloc((size_t)128 * 64 * 2);
    half_t* xh   = (half_t*)alloc((size_t)Nn * 256 * 2);   // x   in f16
    half_t* eah  = (half_t*)alloc((size_t)E * 64 * 2);     // ea  in f16
    half_t* h16  = (half_t*)alloc((size_t)Nn * 128 * 2);   // relu(h1) in f16
    float* qbuf  = (float*)alloc((size_t)Nn * 128 * 4);
    float* kbuf  = (float*)alloc((size_t)Nn * 128 * 4);
    float* vbuf  = (float*)alloc((size_t)Nn * 128 * 4);
    float* hbuf  = (float*)alloc((size_t)Nn * 128 * 4);
    float* ebuf  = (float*)alloc((size_t)E * 128 * 4);
    float* abuf  = (float*)alloc((size_t)E * 4 * 4);
    unsigned* mkey = (unsigned*)alloc((size_t)Nn * 4 * 4);
    float*    ssum = (float*)alloc((size_t)Nn * 4 * 4);
    float* out = (float*)d_out;

    // ---- weight transposes (fp32 -> f16 [N][K]) ----
    auto tpose = [&](const float* w, half_t* wt, int K) {
        transpose_to_f16<<<(K * 128 + 255) / 256, 256, 0, stream>>>(w, wt, K);
    };
    tpose(wq1, wq1T, 256); tpose(wk1, wk1T, 256); tpose(wv1, wv1T, 256); tpose(ws1, ws1T, 256);
    tpose(we1, we1T, 64);
    tpose(wq2, wq2T, 128); tpose(wk2, wk2T, 128); tpose(wv2, wv2T, 128); tpose(ws2, ws2T, 128);
    tpose(we2, we2T, 64);

    // ---- activation conversions to f16 ----
    {
        long nx = (long)Nn * 256;
        f32_to_f16<<<(int)((nx / 4 + 255) / 256), 256, 0, stream>>>(x, xh, nx);
        long ne = (long)E * 64;
        f32_to_f16<<<(int)((ne / 4 + 255) / 256), 256, 0, stream>>>(ea, eah, ne);
    }

    const int gN = (Nn + BM - 1) / BM;
    const int gE = (E  + BM - 1) / BM;
    const int tEH = (E * 4 + 255) / 256;
    const int tNH = (Nn * 4 + 255) / 256;

    // ================= layer 1 =================
    gemm128_wmma_async<<<gN, 256, 0, stream>>>(xh, wq1T, bq1, qbuf, Nn, 256);
    gemm128_wmma_async<<<gN, 256, 0, stream>>>(xh, wk1T, bk1, kbuf, Nn, 256);
    gemm128_wmma_async<<<gN, 256, 0, stream>>>(xh, wv1T, bv1, vbuf, Nn, 256);
    gemm128_wmma_async<<<gN, 256, 0, stream>>>(xh, ws1T, bs1, hbuf, Nn, 256);   // skip path
    gemm128_wmma_async<<<gE, 256, 0, stream>>>(eah, we1T, nullptr, ebuf, E, 64);

    init_stats<<<tNH, 256, 0, stream>>>(mkey, ssum, Nn * 4);
    alpha_kernel<<<tEH, 256, 0, stream>>>(srcI, dstI, qbuf, kbuf, ebuf, abuf, mkey, E);
    expsum_kernel<<<tEH, 256, 0, stream>>>(dstI, abuf, mkey, ssum, E);
    aggregate_kernel<<<tEH, 256, 0, stream>>>(srcI, dstI, vbuf, ebuf, abuf, ssum, hbuf, E);
    {
        long nh = (long)Nn * 128;
        relu_to_f16<<<(int)((nh / 4 + 255) / 256), 256, 0, stream>>>(hbuf, h16, nh);
    }

    // ================= layer 2 =================
    gemm128_wmma_async<<<gN, 256, 0, stream>>>(h16, wq2T, bq2, qbuf, Nn, 128);
    gemm128_wmma_async<<<gN, 256, 0, stream>>>(h16, wk2T, bk2, kbuf, Nn, 128);
    gemm128_wmma_async<<<gN, 256, 0, stream>>>(h16, wv2T, bv2, vbuf, Nn, 128);
    gemm128_wmma_async<<<gN, 256, 0, stream>>>(h16, ws2T, bs2, out, Nn, 128);   // skip -> d_out
    gemm128_wmma_async<<<gE, 256, 0, stream>>>(eah, we2T, nullptr, ebuf, E, 64);

    init_stats<<<tNH, 256, 0, stream>>>(mkey, ssum, Nn * 4);
    alpha_kernel<<<tEH, 256, 0, stream>>>(srcI, dstI, qbuf, kbuf, ebuf, abuf, mkey, E);
    expsum_kernel<<<tEH, 256, 0, stream>>>(dstI, abuf, mkey, ssum, E);
    aggregate_kernel<<<tEH, 256, 0, stream>>>(srcI, dstI, vbuf, ebuf, abuf, ssum, out, E);
}
